// BPaCoLoss_28698971472101
// MI455X (gfx1250) — matrix operand order repaired and checked by reading, not compiled, not run on hardware
//
#include <hip/hip_runtime.h>
#include <hip/hip_bf16.h>
#include <hip/hip_fp16.h>
#include <math.h>
#include <stdint.h>

// ---------------- problem constants (from reference) ----------------
#define BATCH       4096
#define NUM_CLASSES 1000
#define C_PAD       1024      // padded class dim for 64-wide GEMM tiles
#define EMBED       512
#define QUEUE       8192
#define INV_TEMP    (1.0f / 0.07f)
#define CLS_W       1.0f
#define CON_W       0.1f
#define EPS_PRIOR   1e-8f

typedef __attribute__((ext_vector_type(16))) __bf16 bf16x16;
typedef __attribute__((ext_vector_type(8)))  __bf16 bf16x8;
typedef __attribute__((ext_vector_type(8)))  float  f32x8;

// ---------------- workspace layout (bytes, 256-aligned) ----------------
static constexpr size_t OFF_FEATB   = 0;
static constexpr size_t OFF_QUEUEB  = OFF_FEATB  + (size_t)BATCH * EMBED * 2;
static constexpr size_t OFF_CENTB   = OFF_QUEUEB + (size_t)QUEUE * EMBED * 2;
static constexpr size_t OFF_SIMT    = OFF_CENTB  + (size_t)C_PAD * EMBED * 2;
static constexpr size_t OFF_CENTERT = OFF_SIMT   + (size_t)QUEUE * BATCH * 2;
static constexpr size_t OFF_QLSET   = OFF_CENTERT+ (size_t)C_PAD * BATCH * 4;
static constexpr size_t OFF_PA      = OFF_QLSET  + (size_t)C_PAD * BATCH * 4;
static constexpr size_t OFF_COUNTS  = OFF_PA     + 4096;
static constexpr size_t OFF_CURSOR  = OFF_COUNTS + 4096;
static constexpr size_t OFF_OFFSETS = OFF_CURSOR + 4096;
static constexpr size_t OFF_SORTED  = OFF_OFFSETS+ 4096;

// ---------------- small helper kernels ----------------

__global__ void init_misc(const float* __restrict__ prior, float* __restrict__ pa,
                          int* __restrict__ counts, int* __restrict__ cursor,
                          float* __restrict__ out) {
  int i = threadIdx.x;
  counts[i] = 0;
  cursor[i] = 0;
  if (i < NUM_CLASSES) pa[i] = logf(fmaxf(prior[i], EPS_PRIOR));
  if (i == 0) out[0] = 0.0f;
}

__global__ void l2norm_to_bf16(const float* __restrict__ x, __bf16* __restrict__ out,
                               int valid_rows) {
  const int r = blockIdx.x;
  const int tid = threadIdx.x;           // blockDim = 128
  if (r >= valid_rows) {
    for (int i = tid; i < EMBED; i += 128) out[(size_t)r * EMBED + i] = (__bf16)0.0f;
    return;
  }
  __shared__ float red[128];
  __shared__ float sc;
  float ss = 0.0f;
  for (int i = tid; i < EMBED; i += 128) {
    float v = x[(size_t)r * EMBED + i];
    ss += v * v;
  }
  red[tid] = ss;
  __syncthreads();
  for (int w = 64; w > 0; w >>= 1) {
    if (tid < w) red[tid] += red[tid + w];
    __syncthreads();
  }
  if (tid == 0) sc = 1.0f / fmaxf(sqrtf(red[0]), 1e-12f);
  __syncthreads();
  for (int i = tid; i < EMBED; i += 128)
    out[(size_t)r * EMBED + i] = (__bf16)(x[(size_t)r * EMBED + i] * sc);
}

__global__ void to_bf16(const float* __restrict__ x, __bf16* __restrict__ out, int n) {
  int i = blockIdx.x * blockDim.x + threadIdx.x;
  if (i < n) out[i] = (__bf16)x[i];
}

__global__ void hist_labels(const int* __restrict__ labels, int* __restrict__ counts) {
  int q = blockIdx.x * blockDim.x + threadIdx.x;
  if (q < QUEUE) {
    int lab = labels[q];
    if (lab >= 0 && lab < NUM_CLASSES) atomicAdd(&counts[lab], 1);
  }
}

__global__ void scan_counts(const int* __restrict__ counts, int* __restrict__ offsets) {
  if (threadIdx.x == 0) {
    int acc = 0;
    for (int c = 0; c < NUM_CLASSES; ++c) { offsets[c] = acc; acc += counts[c]; }
  }
}

__global__ void scatter_sorted(const int* __restrict__ labels, const int* __restrict__ offsets,
                               int* __restrict__ cursor, int* __restrict__ sorted) {
  int q = blockIdx.x * blockDim.x + threadIdx.x;
  if (q < QUEUE) {
    int lab = labels[q];
    if (lab >= 0 && lab < NUM_CLASSES) {
      int pos = atomicAdd(&cursor[lab], 1);
      sorted[offsets[lab] + pos] = q;
    }
  }
}

// ---------------- WMMA GEMM with async-LDS B staging ----------------
// out[m,n] = scale * sum_k A[m,k] * F[n,k]
// A: bf16 [Mtot x 512] row-major (per-wave 16 M rows, fragments from global/L2),
// F: bf16 [4096 x 512] row-major; 64x32 B tile staged in LDS per block via
// GLOBAL_LOAD_ASYNC_TO_LDS_B128 (ASYNCcnt), double buffered, shared by 4 waves.

#define LDS_ROW 40   // 32 bf16 data + 8 bf16 pad (80 B) -> breaks 16-bank stride

__device__ inline bf16x16 load_afrag(const __bf16* __restrict__ base, int row, int k0, int h) {
  const __bf16* p = base + (size_t)row * EMBED + k0 + h * 8;
  bf16x8 lo = *(const bf16x8*)(p);        // K = k0 + 8h .. +7
  bf16x8 hi = *(const bf16x8*)(p + 16);   // K = k0 + 8h + 16 .. +23
  bf16x16 r;
#pragma unroll
  for (int i = 0; i < 8; ++i) { r[i] = lo[i]; r[8 + i] = hi[i]; }
  return r;
}

__device__ inline bf16x16 load_bfrag_lds(const __bf16* base, int ln, int h) {
  // dense 16-bit B layout: lane holds K = 8h..8h+7 and 16+8h..16+8h+7 for column n
  const __bf16* p = base + ln * LDS_ROW + h * 8;
  bf16x8 lo = *(const bf16x8*)(p);
  bf16x8 hi = *(const bf16x8*)(p + 16);
  bf16x16 r;
#pragma unroll
  for (int i = 0; i < 8; ++i) { r[i] = lo[i]; r[8 + i] = hi[i]; }
  return r;
}

// copy F[n0..n0+63, k0..k0+31] -> LDS tile (rows padded to LDS_ROW elements)
__device__ inline void async_copy_btile(const __bf16* __restrict__ F, int n0, int k0,
                                        __bf16* ldsbase, int tid) {
#pragma unroll
  for (int c = tid; c < 256; c += 128) {        // 256 chunks of 16 B, 2 per thread
    const int row = c >> 2;                     // 0..63
    const int col = c & 3;                      // 16-B chunk within 64-B row
    const __bf16* gp = F + (size_t)(n0 + row) * EMBED + k0 + col * 8;
    // generic shared pointer: low 32 bits are the LDS byte offset (ISA aperture rule)
    unsigned lds_off = (unsigned)(uintptr_t)(ldsbase + row * LDS_ROW + col * 8);
    unsigned long long ga = (unsigned long long)(uintptr_t)gp;
    asm volatile("global_load_async_to_lds_b128 %0, %1, off"
                 :: "v"(lds_off), "v"(ga) : "memory");
  }
}

template <typename OutT>
__global__ void wmma_gemm_nt(const __bf16* __restrict__ A, const __bf16* __restrict__ F,
                             OutT* __restrict__ out, int ldo, float scale,
                             const int* __restrict__ init_mask) {
  __shared__ __bf16 Btile[2][64 * LDS_ROW];   // double-buffered 64x32 bf16 tile

  const int tid  = threadIdx.x;               // blockDim = 128 (4 waves)
  const int wave = tid >> 5;
  const int lane = tid & 31;
  const int row  = lane & 15;
  const int h    = lane >> 4;                 // half-wave select
  const int m0   = (blockIdx.x * 4 + wave) * 16;
  const int n0   = blockIdx.y * 64;

  f32x8 acc0 = {}, acc1 = {}, acc2 = {}, acc3 = {};

  async_copy_btile(F, n0, 0, &Btile[0][0], tid);

  for (int kk = 0; kk < EMBED / 32; ++kk) {
    asm volatile("s_wait_asynccnt 0x0" ::: "memory");  // own copies landed in LDS
    __syncthreads();                                   // whole tile visible to block
    const int cur = kk & 1;
    if (kk + 1 < EMBED / 32)
      async_copy_btile(F, n0, (kk + 1) * 32, &Btile[(kk + 1) & 1][0], tid);

    bf16x16 a  = load_afrag(A, m0 + row, kk * 32, h);
    bf16x16 b0 = load_bfrag_lds(&Btile[cur][0],  0 + row, h);
    bf16x16 b1 = load_bfrag_lds(&Btile[cur][0], 16 + row, h);
    bf16x16 b2 = load_bfrag_lds(&Btile[cur][0], 32 + row, h);
    bf16x16 b3 = load_bfrag_lds(&Btile[cur][0], 48 + row, h);
    acc0 = __builtin_amdgcn_wmma_f32_16x16x32_bf16(false, a, false, b0, (short)0, acc0, false, false);
    acc1 = __builtin_amdgcn_wmma_f32_16x16x32_bf16(false, a, false, b1, (short)0, acc1, false, false);
    acc2 = __builtin_amdgcn_wmma_f32_16x16x32_bf16(false, a, false, b2, (short)0, acc2, false, false);
    acc3 = __builtin_amdgcn_wmma_f32_16x16x32_bf16(false, a, false, b3, (short)0, acc3, false, false);
    // no trailing barrier: top-of-loop wait+barrier protects buffer reuse
  }

  // D layout: lanes 0-15 -> N=lane, M=vgpr r; lanes 16-31 -> N=lane-16, M=r+8
  f32x8 accs[4] = {acc0, acc1, acc2, acc3};
#pragma unroll
  for (int t = 0; t < 4; ++t) {
    const int nn = n0 + t * 16 + (lane & 15);
#pragma unroll
    for (int r = 0; r < 8; ++r) {
      const int mm = m0 + r + 8 * h;
      float v = accs[t][r] * scale;
      if (init_mask) {
        bool keep = (mm < NUM_CLASSES) && (init_mask[mm] != 0);
        if (!keep) v = 0.0f;
      }
      out[(size_t)mm * ldo + nn] = (OutT)v;
    }
  }
}

// ---------------- class-averaged queue log-sum-exp ----------------
__global__ void seg_lse(const _Float16* __restrict__ simT, const int* __restrict__ sorted,
                        const int* __restrict__ offsets, const int* __restrict__ counts,
                        float* __restrict__ qlseT) {
  const int c = blockIdx.y;
  const int b = blockIdx.x * blockDim.x + threadIdx.x;
  const int cnt = counts[c];
  float outv = 0.0f;                       // empty class -> 0 (matches reference agg)
  if (cnt > 0) {
    const int off = offsets[c];
    float m = -INFINITY;
    for (int k = 0; k < cnt; ++k) {
      int j = sorted[off + k];
      float v = (float)simT[(size_t)j * BATCH + b];
      m = fmaxf(m, v);
    }
    float s = 0.0f;
    for (int k = 0; k < cnt; ++k) {
      int j = sorted[off + k];
      float v = (float)simT[(size_t)j * BATCH + b];
      s += expf(v - m);
    }
    outv = m + logf(s) - logf((float)cnt);
  }
  qlseT[(size_t)c * BATCH + b] = outv;
}

// ---------------- cross-entropy terms ----------------

__device__ inline void lse_push(float v, float& m, float& s) {
  if (v <= m) { s += expf(v - m); }
  else        { s = s * expf(m - v) + 1.0f; m = v; }
}

__global__ void cls_ce(const float* __restrict__ logits, const int* __restrict__ targets,
                       const float* __restrict__ pa, float* __restrict__ out) {
  const int b = blockIdx.x;
  const int tid = threadIdx.x;             // blockDim = 128
  const float* rowp = logits + (size_t)b * NUM_CLASSES;
  float m = -INFINITY, s = 0.0f;
  for (int c = tid; c < NUM_CLASSES; c += 128) lse_push(rowp[c] - pa[c], m, s);
  __shared__ float sm[128], ss[128];
  sm[tid] = m; ss[tid] = s;
  __syncthreads();
  for (int w = 64; w > 0; w >>= 1) {
    if (tid < w) {
      float m1 = sm[tid], s1 = ss[tid], m2 = sm[tid + w], s2 = ss[tid + w];
      float M = fmaxf(m1, m2);
      sm[tid] = M;
      ss[tid] = s1 * expf(m1 - M) + s2 * expf(m2 - M);
    }
    __syncthreads();
  }
  if (tid == 0) {
    float lse = sm[0] + logf(ss[0]);
    int t = targets[b];
    float nll = lse - (rowp[t] - pa[t]);
    atomicAdd(out, nll * (CLS_W / (float)BATCH));
  }
}

__device__ inline float logaddexpf(float a, float b) {
  float M = fmaxf(a, b);
  return M + log1pf(expf(fminf(a, b) - M));
}

__global__ void con_ce(const float* __restrict__ centerT, const float* __restrict__ qlseT,
                       const int* __restrict__ targets, const float* __restrict__ pa,
                       float* __restrict__ out) {
  const int b = blockIdx.x * blockDim.x + threadIdx.x;   // 0..4095
  float m = -INFINITY, s = 0.0f;
  for (int c = 0; c < NUM_CLASSES; ++c) {
    float comp = logaddexpf(centerT[(size_t)c * BATCH + b], qlseT[(size_t)c * BATCH + b]);
    lse_push(comp - pa[c], m, s);
  }
  int t = targets[b];
  float compT = logaddexpf(centerT[(size_t)t * BATCH + b], qlseT[(size_t)t * BATCH + b]);
  float nll = (m + logf(s)) - (compT - pa[t]);
  atomicAdd(out, nll * (CON_W / (float)BATCH));
}

// ---------------- host launcher ----------------
extern "C" void kernel_launch(void* const* d_in, const int* in_sizes, int n_in,
                              void* d_out, int out_size, void* d_ws, size_t ws_size,
                              hipStream_t stream) {
  (void)in_sizes; (void)n_in; (void)out_size; (void)ws_size;

  const float* logits      = (const float*)d_in[0];  // [4096,1000]
  const float* embeddings  = (const float*)d_in[1];  // [4096,512]
  const float* centers     = (const float*)d_in[2];  // [1000,512]
  const float* queue       = (const float*)d_in[3];  // [8192,512]
  const float* prior       = (const float*)d_in[4];  // [1000]
  const int*   targets     = (const int*)d_in[5];    // [4096]
  const int*   init_mask   = (const int*)d_in[6];    // [1000] (bool as int)
  const int*   qlabels     = (const int*)d_in[7];    // [8192]
  float*       out         = (float*)d_out;

  char* ws = (char*)d_ws;
  __bf16*    featB   = (__bf16*)(ws + OFF_FEATB);
  __bf16*    queueB  = (__bf16*)(ws + OFF_QUEUEB);
  __bf16*    centB   = (__bf16*)(ws + OFF_CENTB);
  _Float16*  simT    = (_Float16*)(ws + OFF_SIMT);
  float*     centerT = (float*)(ws + OFF_CENTERT);
  float*     qlseT   = (float*)(ws + OFF_QLSET);
  float*     pa      = (float*)(ws + OFF_PA);
  int*       counts  = (int*)(ws + OFF_COUNTS);
  int*       cursor  = (int*)(ws + OFF_CURSOR);
  int*       offsets = (int*)(ws + OFF_OFFSETS);
  int*       sorted  = (int*)(ws + OFF_SORTED);

  // stage 0: counters / prior adjust / loss = 0
  init_misc<<<1, 1024, 0, stream>>>(prior, pa, counts, cursor, out);

  // stage 1: normalize + bf16 convert
  l2norm_to_bf16<<<BATCH, 128, 0, stream>>>(embeddings, featB, BATCH);
  l2norm_to_bf16<<<C_PAD, 128, 0, stream>>>(centers, centB, NUM_CLASSES);
  to_bf16<<<(QUEUE * EMBED) / 256, 256, 0, stream>>>(queue, queueB, QUEUE * EMBED);

  // stage 2: counting-sort queue indices by class
  hist_labels<<<QUEUE / 256, 256, 0, stream>>>(qlabels, counts);
  scan_counts<<<1, 32, 0, stream>>>(counts, offsets);
  scatter_sorted<<<QUEUE / 256, 256, 0, stream>>>(qlabels, offsets, cursor, sorted);

  // stage 3: WMMA GEMMs (outputs transposed: [M, B])
  {
    dim3 grid(C_PAD / 64, BATCH / 64);
    wmma_gemm_nt<float><<<grid, 128, 0, stream>>>(centB, featB, centerT, BATCH,
                                                  INV_TEMP, init_mask);
  }
  {
    dim3 grid(QUEUE / 64, BATCH / 64);
    wmma_gemm_nt<_Float16><<<grid, 128, 0, stream>>>(queueB, featB, simT, BATCH,
                                                     INV_TEMP, nullptr);
  }

  // stage 4: class-averaged queue log-sum-exp  -> qlseT [C,B]
  {
    dim3 grid(BATCH / 256, NUM_CLASSES);
    seg_lse<<<grid, 256, 0, stream>>>(simT, sorted, offsets, counts, qlseT);
  }

  // stage 5: the two cross-entropy terms, accumulated into out[0]
  cls_ce<<<BATCH, 128, 0, stream>>>(logits, targets, pa, out);
  con_ce<<<BATCH / 256, 256, 0, stream>>>(centerT, qlseT, targets, pa, out);
}
